// RNNModel_68899865362706
// MI455X (gfx1250) — matrix operand order repaired and checked by reading, not compiled, not run on hardware
//
#include <hip/hip_runtime.h>

// ---------------------------------------------------------------------------
// Fused 3-layer tanh-RNN + FC head for MI455X (gfx1250, wave32, WMMA).
//   B=32768, T=60, H=64, D_FEAT=6.
// One wave owns a 16-row batch tile; per timestep it runs all three layers
// (per-timestep layer fusion) with v_wmma_f32_16x16x32_bf16, hardware
// v_tanh_f32 activations, and an LDS round-trip for C-layout -> A-layout.
// ---------------------------------------------------------------------------

#define BATCH          32768
#define TSTEPS         60
#define DFEAT          6
#define HDIM           64
#define BLOCK_THREADS  256
#define WAVES_PER_BLK  8
#define ROWS_PER_WAVE  16
#define ROWS_PER_BLK   (WAVES_PER_BLK * ROWS_PER_WAVE)

typedef __bf16 bf16_t;
typedef bf16_t v16bf __attribute__((ext_vector_type(16)));
typedef float  v8f   __attribute__((ext_vector_type(8)));
typedef unsigned int u32x4 __attribute__((ext_vector_type(4)));

union V16U { v16bf v; u32x4 q[2]; };

__device__ __forceinline__ float fast_tanh(float x) {
#if __has_builtin(__builtin_amdgcn_tanhf)
  return __builtin_amdgcn_tanhf(x);           // v_tanh_f32 (TRANS32)
#else
  // branchless fallback: tanh(x) = 1 - 2/(exp(2x)+1)
  float e = __builtin_amdgcn_exp2f(2.8853900817779268f * x);   // e^(2x)
  return 1.0f - 2.0f * __builtin_amdgcn_rcpf(e + 1.0f);
#endif
}

__device__ __forceinline__ v8f wmma_bf16(v16bf a, v16bf b, v8f c) {
  // (neg_a, A, neg_b, B, c_mod, C, reuse_a, reuse_b)
  return __builtin_amdgcn_wmma_f32_16x16x32_bf16(false, a, false, b,
                                                 (short)0, c, false, false);
}

// B operand (32x16 bf16): B[k][n] = W[n][k]; W staged row-major in LDS.
// lane L (0..15):  n = nt*16+L, K = kb*32 + 0..15  (16 consecutive halves)
// lane L+16:       n = nt*16+L, K = kb*32 + 16..31
__device__ __forceinline__ v16bf load_b(const bf16_t* w, int rowstride,
                                        int nt, int kb, int lane) {
  int n  = nt * 16 + (lane & 15);
  int k0 = kb * 32 + ((lane >> 4) << 4);
  const bf16_t* p = w + n * rowstride + k0;
  V16U u;
  u.q[0] = *(const u32x4*)(p);
  u.q[1] = *(const u32x4*)(p + 8);
  return u.v;
}

// A operand (16x32 bf16) from row-major staged [16][64] tile.
// lane L (0..15):  m = L, K = kb*32 + 0..15
// lane L+16:       m = L, K = kb*32 + 16..31
__device__ __forceinline__ v16bf load_a(const bf16_t* hst, int kb, int lane) {
  int m  = lane & 15;
  int k0 = kb * 32 + ((lane >> 4) << 4);
  const bf16_t* p = hst + m * HDIM + k0;
  V16U u;
  u.q[0] = *(const u32x4*)(p);
  u.q[1] = *(const u32x4*)(p + 8);
  return u.v;
}

// C/D layout (16x16 f32): VGPR v, lane L -> (M = v + 8*(L>=16), N = L%16).
__device__ __forceinline__ void store_c(bf16_t* hst, int nt, int lane, v8f c) {
  int n  = nt * 16 + (lane & 15);
  int m0 = (lane >> 4) << 3;
#pragma unroll
  for (int v = 0; v < 8; ++v) hst[(m0 + v) * HDIM + n] = (bf16_t)c[v];
}

// Interleave the 4 independent N-tile accumulators so the scheduler always
// has 4 independent WMMAs in flight between dependent (D->C) pairs.
__device__ __forceinline__ void accum4(v8f c[4], v16bf a, const bf16_t* w,
                                       int rowstride, int kb, int lane) {
#pragma unroll
  for (int nt = 0; nt < 4; ++nt)
    c[nt] = wmma_bf16(a, load_b(w, rowstride, nt, kb, lane), c[nt]);
}

__global__ __launch_bounds__(BLOCK_THREADS)
void rnn3_fused_wmma(const float* __restrict__ x,
                     const float* __restrict__ wih0, const float* __restrict__ whh0,
                     const float* __restrict__ bih0, const float* __restrict__ bhh0,
                     const float* __restrict__ wih1, const float* __restrict__ whh1,
                     const float* __restrict__ bih1, const float* __restrict__ bhh1,
                     const float* __restrict__ wih2, const float* __restrict__ whh2,
                     const float* __restrict__ bih2, const float* __restrict__ bhh2,
                     const float* __restrict__ fcw,  const float* __restrict__ fcb,
                     float* __restrict__ out) {
  // 0:Whh0 1:Wih1 2:Whh1 3:Wih2 4:Whh2 (each 64x64 bf16, row-major)
  __shared__ bf16_t sW[5][HDIM * HDIM];
  __shared__ bf16_t sWih0[HDIM * 32];   // 64x6 zero-padded to 64x32
  __shared__ float  sBias[3][HDIM];     // b_ih + b_hh per layer
  __shared__ float  sFcw[HDIM];
  __shared__ float  sFcb;
  __shared__ bf16_t sStage[WAVES_PER_BLK][ROWS_PER_WAVE * HDIM];

  const int tid = threadIdx.x;
  {
    const float* wsrc[5] = { whh0, wih1, whh1, wih2, whh2 };
#pragma unroll
    for (int w = 0; w < 5; ++w)
      for (int i = tid; i < HDIM * HDIM; i += BLOCK_THREADS)
        sW[w][i] = (bf16_t)wsrc[w][i];
    for (int i = tid; i < HDIM * 32; i += BLOCK_THREADS) {
      int n = i >> 5, k = i & 31;
      sWih0[i] = (k < DFEAT) ? (bf16_t)wih0[n * DFEAT + k] : (bf16_t)0.0f;
    }
    for (int i = tid; i < HDIM; i += BLOCK_THREADS) {
      sBias[0][i] = bih0[i] + bhh0[i];
      sBias[1][i] = bih1[i] + bhh1[i];
      sBias[2][i] = bih2[i] + bhh2[i];
      sFcw[i] = fcw[i];
    }
    if (tid == 0) sFcb = fcb[0];
  }
  __syncthreads();

  const int lane = tid & 31;
  const int wid  = tid >> 5;
  const int r0   = blockIdx.x * ROWS_PER_BLK + wid * ROWS_PER_WAVE;
  bf16_t* stage = sStage[wid];

  V16U zz; zz.q[0] = (u32x4)0; zz.q[1] = (u32x4)0;
  v16bf hA0_0 = zz.v, hA0_1 = zz.v;   // h_l in WMMA A layout (two 16x32 chunks)
  v16bf hA1_0 = zz.v, hA1_1 = zz.v;
  v16bf hA2_0 = zz.v, hA2_1 = zz.v;

  float bias0[4], bias1[4], bias2[4];
#pragma unroll
  for (int nt = 0; nt < 4; ++nt) {
    bias0[nt] = sBias[0][nt * 16 + (lane & 15)];
    bias1[nt] = sBias[1][nt * 16 + (lane & 15)];
    bias2[nt] = sBias[2][nt * 16 + (lane & 15)];
  }

#pragma unroll 1
  for (int t = 0; t < TSTEPS; ++t) {
    // Build x_t A-operand (16x32; K = feature dim, 6 real, rest zero).
    // xs[b][t][d] = x[b*360 + d*60 + t]
    // 16-bit A layout: lanes 0-15 hold K=0..7 in elements 0..7 -> d goes there.
    V16U xa; xa.q[0] = (u32x4)0; xa.q[1] = (u32x4)0;
    if (lane < 16) {
      const float* xr = x + (size_t)(r0 + lane) * (DFEAT * TSTEPS) + t;
#pragma unroll
      for (int d = 0; d < DFEAT; ++d) xa.v[d] = (bf16_t)xr[d * TSTEPS];
    }

    // -------- layer 0: h0 = tanh(x_t Wih0^T + b + h0 Whh0^T) --------
    {
      v8f c[4];
#pragma unroll
      for (int nt = 0; nt < 4; ++nt)
#pragma unroll
        for (int i = 0; i < 8; ++i) c[nt][i] = bias0[nt];
      accum4(c, xa.v,  sWih0, 32,  0, lane);
      accum4(c, hA0_0, sW[0], HDIM, 0, lane);
      accum4(c, hA0_1, sW[0], HDIM, 1, lane);
#pragma unroll
      for (int nt = 0; nt < 4; ++nt) {
#pragma unroll
        for (int i = 0; i < 8; ++i) c[nt][i] = fast_tanh(c[nt][i]);
        store_c(stage, nt, lane, c[nt]);
      }
    }
    hA0_0 = load_a(stage, 0, lane);
    hA0_1 = load_a(stage, 1, lane);

    // -------- layer 1: h1 = tanh(h0 Wih1^T + b + h1 Whh1^T) --------
    {
      v8f c[4];
#pragma unroll
      for (int nt = 0; nt < 4; ++nt)
#pragma unroll
        for (int i = 0; i < 8; ++i) c[nt][i] = bias1[nt];
      accum4(c, hA0_0, sW[1], HDIM, 0, lane);
      accum4(c, hA0_1, sW[1], HDIM, 1, lane);
      accum4(c, hA1_0, sW[2], HDIM, 0, lane);
      accum4(c, hA1_1, sW[2], HDIM, 1, lane);
#pragma unroll
      for (int nt = 0; nt < 4; ++nt) {
#pragma unroll
        for (int i = 0; i < 8; ++i) c[nt][i] = fast_tanh(c[nt][i]);
        store_c(stage, nt, lane, c[nt]);
      }
    }
    hA1_0 = load_a(stage, 0, lane);
    hA1_1 = load_a(stage, 1, lane);

    // -------- layer 2: h2 = tanh(h1 Wih2^T + b + h2 Whh2^T) --------
    {
      v8f c[4];
#pragma unroll
      for (int nt = 0; nt < 4; ++nt)
#pragma unroll
        for (int i = 0; i < 8; ++i) c[nt][i] = bias2[nt];
      accum4(c, hA1_0, sW[3], HDIM, 0, lane);
      accum4(c, hA1_1, sW[3], HDIM, 1, lane);
      accum4(c, hA2_0, sW[4], HDIM, 0, lane);
      accum4(c, hA2_1, sW[4], HDIM, 1, lane);
#pragma unroll
      for (int nt = 0; nt < 4; ++nt) {
#pragma unroll
        for (int i = 0; i < 8; ++i) c[nt][i] = fast_tanh(c[nt][i]);
        store_c(stage, nt, lane, c[nt]);
      }
    }
    hA2_0 = load_a(stage, 0, lane);
    hA2_1 = load_a(stage, 1, lane);
  }

  // FC head on h2[T-1] (row-major in this wave's stage buffer).
  if (lane < 16) {
    float s = sFcb;
#pragma unroll
    for (int k = 0; k < HDIM; ++k)
      s += (float)stage[lane * HDIM + k] * sFcw[k];
    out[r0 + lane] = s;
  }
}

extern "C" void kernel_launch(void* const* d_in, const int* in_sizes, int n_in,
                              void* d_out, int out_size, void* d_ws, size_t ws_size,
                              hipStream_t stream) {
  (void)in_sizes; (void)n_in; (void)out_size; (void)d_ws; (void)ws_size;
  const float* x    = (const float*)d_in[0];
  const float* wih0 = (const float*)d_in[1];
  const float* whh0 = (const float*)d_in[2];
  const float* bih0 = (const float*)d_in[3];
  const float* bhh0 = (const float*)d_in[4];
  const float* wih1 = (const float*)d_in[5];
  const float* whh1 = (const float*)d_in[6];
  const float* bih1 = (const float*)d_in[7];
  const float* bhh1 = (const float*)d_in[8];
  const float* wih2 = (const float*)d_in[9];
  const float* whh2 = (const float*)d_in[10];
  const float* bih2 = (const float*)d_in[11];
  const float* bhh2 = (const float*)d_in[12];
  const float* fcw  = (const float*)d_in[13];
  const float* fcb  = (const float*)d_in[14];
  float* out = (float*)d_out;

  dim3 grid(BATCH / ROWS_PER_BLK);   // 256 workgroups x 128 rows = 32768
  dim3 block(BLOCK_THREADS);         // 8 waves, one 16-row tile each
  rnn3_fused_wmma<<<grid, block, 0, stream>>>(
      x, wih0, whh0, bih0, bhh0, wih1, whh1, bih1, bhh1,
      wih2, whh2, bih2, bhh2, fcw, fcb, out);
}
